// AttentiveForest_87531433492897
// MI455X (gfx1250) — matrix operand order repaired and checked by reading, not compiled, not run on hardware
//
#include <hip/hip_runtime.h>
#include <hip/hip_bf16.h>

// ---------------------------------------------------------------------------
// AttentiveForest on MI455X (gfx1250, wave32)
//   GEMM1: f_aug(4096x3072) @ sel^T(3072x1536)  -> p      (38.7 GFLOP, WMMA f16)
//   GEMM2: f_aug(4096x3072) @ att_W(3072x256)   -> logits ( 6.4 GFLOP, WMMA f16)
//   GEMMs: 128x64 block tile, 32x32 wave tile (4 WMMA / K-step / wave),
//          double-buffered LDS filled with GLOBAL_LOAD_ASYNC_TO_LDS_B128
//          (ASYNCcnt-tracked CDNA5 async copy), s_wait_asynccnt + barrier.
//   sparsemax via bisection on tau (root of sum(max(z-tau,0))=1), no sort.
// ---------------------------------------------------------------------------

typedef __attribute__((ext_vector_type(16))) _Float16 v16h;
typedef __attribute__((ext_vector_type(8)))  float    v8f;

#define Bsz   4096
#define Dd    256
#define Kk    6
#define FIN   3072      // 2*D*K
#define HALF_FIN 1536   // D*K
#define Tt    256
#define DEPTH 6
#define NLEAF 64
#define Cc    8

union Frag { uint4 u[2]; v16h h; };

__device__ __forceinline__ float warpSum(float v) {
#pragma unroll
  for (int o = 16; o > 0; o >>= 1) v += __shfl_down(v, o, 32);
  return v;
}
__device__ __forceinline__ float warpMax(float v) {
#pragma unroll
  for (int o = 16; o > 0; o >>= 1) v = fmaxf(v, __shfl_down(v, o, 32));
  return v;
}

// ---- CDNA5 async global->LDS copy (tracked by ASYNCcnt) --------------------
__device__ __forceinline__ unsigned ldsByteOff(const void* p) {
  // generic -> LDS addrspace; ptrtoint of an AS(3) pointer is the raw LDS
  // byte offset expected in the async-load VDST operand.
  return (unsigned)(unsigned long long)(__attribute__((address_space(3))) const void*)p;
}
__device__ __forceinline__ void asyncLdsB128(unsigned lds_off, const void* g) {
  asm volatile("global_load_async_to_lds_b128 %0, %1, off"
               :: "v"(lds_off), "v"((unsigned long long)g)
               : "memory");
}
__device__ __forceinline__ void waitAsync0() {
  asm volatile("s_wait_asynccnt 0x0" ::: "memory");
}

// ---------------------------------------------------------------------------
// 1) sparsemax over each row (len 3072) of sel_logits -> f16 (row-major,
//    rows are exactly the B^T rows needed by GEMM1).
// ---------------------------------------------------------------------------
__global__ __launch_bounds__(256) void sel_sparsemax_kernel(
    const float* __restrict__ Z, _Float16* __restrict__ Oh) {
  __shared__ float zs[FIN];
  __shared__ float red[8];
  const int row = blockIdx.x, tid = threadIdx.x;
  const int wid = tid >> 5, lane = tid & 31;
  const float* z = Z + (size_t)row * FIN;

  float lmax = -3.4e38f;
#pragma unroll
  for (int i = 0; i < 12; ++i) {
    float v = z[tid + i * 256];
    zs[tid + i * 256] = v;
    lmax = fmaxf(lmax, v);
  }
  lmax = warpMax(lmax);
  if (lane == 0) red[wid] = lmax;
  __syncthreads();
  float zmax = red[0];
#pragma unroll
  for (int w = 1; w < 8; ++w) zmax = fmaxf(zmax, red[w]);

  float lo = zmax - 1.0f, hi = zmax;   // g(lo) >= 1 >= g(hi), g decreasing
  for (int it = 0; it < 32; ++it) {
    const float mid = 0.5f * (lo + hi);
    float ls = 0.f;
#pragma unroll
    for (int i = 0; i < 12; ++i) ls += fmaxf(zs[tid + i * 256] - mid, 0.f);
    ls = warpSum(ls);
    __syncthreads();                    // protect red[] (readers done)
    if (lane == 0) red[wid] = ls;
    __syncthreads();
    float s = red[0];
#pragma unroll
    for (int w = 1; w < 8; ++w) s += red[w];
    if (s >= 1.0f) lo = mid; else hi = mid;
  }
  const float tau = 0.5f * (lo + hi);
#pragma unroll
  for (int i = 0; i < 12; ++i)
    Oh[(size_t)row * FIN + tid + i * 256] =
        (_Float16)fmaxf(zs[tid + i * 256] - tau, 0.f);
}

// ---------------------------------------------------------------------------
// 2) f_aug = [sigmoid((x-thr)*exp(log_beta)), 1 - sigmoid(...)] as f16
// ---------------------------------------------------------------------------
__global__ __launch_bounds__(256) void faug_kernel(
    const float* __restrict__ x, const float* __restrict__ thr,
    const float* __restrict__ log_beta, _Float16* __restrict__ Fh) {
  const int i = blockIdx.x * 256 + threadIdx.x;   // over B * HALF_FIN
  const int b = i / HALF_FIN;
  const int j = i - b * HALF_FIN;                 // j = d*K + k
  const int d = j / Kk;
  const float zv = (x[(size_t)b * Dd + d] - thr[j]) * __expf(log_beta[j]);
  const float f = 1.0f / (1.0f + __expf(-zv));
  Fh[(size_t)b * FIN + j]            = (_Float16)f;
  Fh[(size_t)b * FIN + HALF_FIN + j] = (_Float16)(1.0f - f);
}

// ---------------------------------------------------------------------------
// 3) att_W (FIN x T) f32 -> att_Wt (T x FIN) f16
// ---------------------------------------------------------------------------
__global__ __launch_bounds__(256) void attw_t_kernel(
    const float* __restrict__ W, _Float16* __restrict__ Wt) {
  const int i = blockIdx.x * 256 + threadIdx.x;   // over T*FIN (output order)
  const int t = i / FIN;
  const int k = i - t * FIN;
  Wt[i] = (_Float16)W[(size_t)k * Tt + t];
}

// ---------------------------------------------------------------------------
// 4) WMMA GEMM:  C(MxN) = A(MxK,f16,row-major) * Bt(NxK,f16,row-major)^T
//    Block tile 128x64, K-step 32. 8 waves: wm=wid&3 (four 32-row strips),
//    wn=wid>>2 (two 32-col strips); each wave = 32x32 = 4 accumulators.
//    Double-buffered LDS, filled by async global->LDS B128 copies.
//    All dims divide tiles -> no divergence, EXEC all-ones for WMMA.
// ---------------------------------------------------------------------------
__global__ __launch_bounds__(256) void wmma_gemm_nt(
    const _Float16* __restrict__ A, const _Float16* __restrict__ Bt,
    float* __restrict__ C, const float* __restrict__ bias,
    int M, int N, int K) {
  // rows padded to 5 uint4 (32 halves + 16B pad) to spread LDS banks
  __shared__ uint4 As4[2][128 * 5];   // 2 x 10 KB
  __shared__ uint4 Bs4[2][64 * 5];    // 2 x  5 KB

  const int tid  = threadIdx.x;
  const int wid  = tid >> 5, lane = tid & 31;
  const int l    = lane & 15, hi = lane >> 4;
  const int wm   = wid & 3,  wn = wid >> 2;
  const int mBase = blockIdx.y * 128, nBase = blockIdx.x * 64;

  // ---- copy assignments: 3 async B128 per thread per K-slab ----
  const int rA0 = tid >> 2;          // A rows 0..63
  const int rA1 = rA0 + 64;          // A rows 64..127
  const int c4  = tid & 3;           // 16B slot within 32-half slab
  unsigned la0[2], la1[2], lb0[2];
#pragma unroll
  for (int b = 0; b < 2; ++b) {
    la0[b] = ldsByteOff(&As4[b][rA0 * 5 + c4]);
    la1[b] = ldsByteOff(&As4[b][rA1 * 5 + c4]);
    lb0[b] = ldsByteOff(&Bs4[b][rA0 * 5 + c4]);
  }
  const _Float16* gA0 = A + (size_t)(mBase + rA0) * K + c4 * 8;
  const _Float16* gA1 = A + (size_t)(mBase + rA1) * K + c4 * 8;
  const _Float16* gB  = Bt + (size_t)(nBase + rA0) * K + c4 * 8;

  v8f acc00 = {}, acc01 = {}, acc10 = {}, acc11 = {};

  // fragment LDS row indices (in uint4 units)
  const int ra0 = (wm * 32 + l) * 5, ra1 = ra0 + 16 * 5;
  const int rb0 = (wn * 32 + l) * 5, rb1 = rb0 + 16 * 5;

  // prologue: fill buffer 0
  asyncLdsB128(la0[0], gA0);
  asyncLdsB128(la1[0], gA1);
  asyncLdsB128(lb0[0], gB);
  waitAsync0();
  __syncthreads();

  int cur = 0;
  for (int kb = 0; kb < K; kb += 32) {
    // stream next K-slab into the other buffer while we compute
    if (kb + 32 < K) {
      const int nxt = cur ^ 1, ko = kb + 32;
      asyncLdsB128(la0[nxt], gA0 + ko);
      asyncLdsB128(la1[nxt], gA1 + ko);
      asyncLdsB128(lb0[nxt], gB + ko);
    }

    const uint4* As = As4[cur];
    const uint4* Bs = Bs4[cur];
    // A fragments 16x32: lane group hi holds K chunks {hi*8..+7, hi*8+16..+23}
    Frag a0, a1, b0, b1;
    a0.u[0] = As[ra0 + hi];     a0.u[1] = As[ra0 + hi + 2];
    a1.u[0] = As[ra1 + hi];     a1.u[1] = As[ra1 + hi + 2];
    // B fragments 32x16: lane l = column, lane group hi holds K chunks {hi*8..+7, hi*8+16..+23}
    b0.u[0] = Bs[rb0 + hi]; b0.u[1] = Bs[rb0 + hi + 2];
    b1.u[0] = Bs[rb1 + hi]; b1.u[1] = Bs[rb1 + hi + 2];

    acc00 = __builtin_amdgcn_wmma_f32_16x16x32_f16(
        false, a0.h, false, b0.h, (short)0, acc00, false, false);
    acc01 = __builtin_amdgcn_wmma_f32_16x16x32_f16(
        false, a0.h, false, b1.h, (short)0, acc01, false, false);
    acc10 = __builtin_amdgcn_wmma_f32_16x16x32_f16(
        false, a1.h, false, b0.h, (short)0, acc10, false, false);
    acc11 = __builtin_amdgcn_wmma_f32_16x16x32_f16(
        false, a1.h, false, b1.h, (short)0, acc11, false, false);

    waitAsync0();        // own async writes to 'nxt' landed in LDS
    __syncthreads();     // everyone done reading 'cur' & writing 'nxt'
    cur ^= 1;
  }

  const int col0 = nBase + wn * 32 + l;
  const int col1 = col0 + 16;
  const float bi0 = bias ? bias[col0] : 0.f;
  const float bi1 = bias ? bias[col1] : 0.f;
  const int row0 = mBase + wm * 32 + hi * 8;   // a0 strip
  const int row1 = row0 + 16;                  // a1 strip
#pragma unroll
  for (int rr = 0; rr < 8; ++rr) {
    C[(size_t)(row0 + rr) * N + col0] = acc00[rr] + bi0;
    C[(size_t)(row0 + rr) * N + col1] = acc01[rr] + bi1;
    C[(size_t)(row1 + rr) * N + col0] = acc10[rr] + bi0;
    C[(size_t)(row1 + rr) * N + col1] = acc11[rr] + bi1;
  }
}

// ---------------------------------------------------------------------------
// 5) sparsemax over T=256 attention logits per batch row (in-place)
// ---------------------------------------------------------------------------
__global__ __launch_bounds__(256) void att_sparsemax_kernel(float* __restrict__ L) {
  __shared__ float red[8];
  const int row = blockIdx.x, tid = threadIdx.x;
  const int wid = tid >> 5, lane = tid & 31;
  const float v = L[(size_t)row * Tt + tid];

  float lmax = warpMax(v);
  if (lane == 0) red[wid] = lmax;
  __syncthreads();
  float zmax = red[0];
#pragma unroll
  for (int w = 1; w < 8; ++w) zmax = fmaxf(zmax, red[w]);

  float lo = zmax - 1.0f, hi = zmax;
  for (int it = 0; it < 32; ++it) {
    const float mid = 0.5f * (lo + hi);
    float ls = warpSum(fmaxf(v - mid, 0.f));
    __syncthreads();
    if (lane == 0) red[wid] = ls;
    __syncthreads();
    float s = red[0];
#pragma unroll
    for (int w = 1; w < 8; ++w) s += red[w];
    if (s >= 1.0f) lo = mid; else hi = mid;
  }
  const float tau = 0.5f * (lo + hi);
  L[(size_t)row * Tt + tid] = fmaxf(v - tau, 0.f);
}

// ---------------------------------------------------------------------------
// 6) soft-tree leaf contraction + attention-weighted sum.
//    thread = tree t; 4 batch rows per block; leaf prob = prod over 6 bits
//    (bit d of leaf selects p_d vs 1-p_d). Deterministic shuffle+LDS reduce.
// ---------------------------------------------------------------------------
__global__ __launch_bounds__(256) void forest_final_kernel(
    const float* __restrict__ P, const float* __restrict__ Aatt,
    const float* __restrict__ leaf, float* __restrict__ out) {
  const int t = threadIdx.x;
  const int bBase = blockIdx.x * 4;
  float pd[4][6], qd[4][6], av[4];
#pragma unroll
  for (int rb = 0; rb < 4; ++rb) {
    const int b = bBase + rb;
    av[rb] = Aatt[(size_t)b * Tt + t];
#pragma unroll
    for (int d = 0; d < 6; ++d) {
      float v = P[(size_t)b * (Tt * DEPTH) + t * DEPTH + d];
      v = fminf(fmaxf(v, 1e-6f), 1.0f - 1e-6f);
      pd[rb][d] = v; qd[rb][d] = 1.0f - v;
    }
  }
  float acc[4][8];
#pragma unroll
  for (int rb = 0; rb < 4; ++rb)
#pragma unroll
    for (int c = 0; c < 8; ++c) acc[rb][c] = 0.f;

  const float* lvt = leaf + (size_t)t * (NLEAF * Cc);
  for (int lf = 0; lf < NLEAF; ++lf) {
    float lv[8];
#pragma unroll
    for (int c = 0; c < 8; ++c) lv[c] = lvt[lf * 8 + c];
#pragma unroll
    for (int rb = 0; rb < 4; ++rb) {
      float pr = ((lf & 1)  ? pd[rb][0] : qd[rb][0]);
      pr      *= ((lf & 2)  ? pd[rb][1] : qd[rb][1]);
      pr      *= ((lf & 4)  ? pd[rb][2] : qd[rb][2]);
      pr      *= ((lf & 8)  ? pd[rb][3] : qd[rb][3]);
      pr      *= ((lf & 16) ? pd[rb][4] : qd[rb][4]);
      pr      *= ((lf & 32) ? pd[rb][5] : qd[rb][5]);
#pragma unroll
      for (int c = 0; c < 8; ++c) acc[rb][c] = fmaf(pr, lv[c], acc[rb][c]);
    }
  }

  __shared__ float red[8][4][8];
  const int wid = t >> 5, lane = t & 31;
#pragma unroll
  for (int rb = 0; rb < 4; ++rb)
#pragma unroll
    for (int c = 0; c < 8; ++c) {
      float v = warpSum(av[rb] * acc[rb][c]);
      if (lane == 0) red[wid][rb][c] = v;
    }
  __syncthreads();
  if (t < 32) {
    const int rb = t >> 3, c = t & 7;
    float s = 0.f;
#pragma unroll
    for (int w = 0; w < 8; ++w) s += red[w][rb][c];
    out[(size_t)(bBase + rb) * Cc + c] = s;
  }
}

// ---------------------------------------------------------------------------
// launch
// ---------------------------------------------------------------------------
extern "C" void kernel_launch(void* const* d_in, const int* in_sizes, int n_in,
                              void* d_out, int out_size, void* d_ws, size_t ws_size,
                              hipStream_t stream) {
  const float* x          = (const float*)d_in[0];  // (4096, 256)
  const float* thr        = (const float*)d_in[1];  // (256, 6)
  const float* log_beta   = (const float*)d_in[2];  // (256, 6)
  const float* sel_logits = (const float*)d_in[3];  // (1536, 3072)
  const float* leaf       = (const float*)d_in[4];  // (256, 64, 8)
  const float* attW       = (const float*)d_in[5];  // (3072, 256)
  const float* attb       = (const float*)d_in[6];  // (256,)
  float* out = (float*)d_out;                       // (4096, 8)

  char* ws = (char*)d_ws;
  // workspace layout (all regions fully rewritten every call):
  _Float16* selH   = (_Float16*)(ws + 0);                 //  9,437,184 B
  _Float16* faugH  = (_Float16*)(ws + 9437184);           // 25,165,824 B
  _Float16* attWtH = (_Float16*)(ws + 34603008);          //  1,572,864 B
  float*    Pp     = (float*)   (ws + 36175872);          // 25,165,824 B
  float*    Aatt   = (float*)   (ws + 61341696);          //  4,194,304 B

  sel_sparsemax_kernel<<<Tt * DEPTH, 256, 0, stream>>>(sel_logits, selH);
  faug_kernel<<<(Bsz * HALF_FIN) / 256, 256, 0, stream>>>(x, thr, log_beta, faugH);
  attw_t_kernel<<<(Tt * FIN) / 256, 256, 0, stream>>>(attW, attWtH);

  // GEMM1: p = f_aug @ sel^T   (M=4096, N=1536, K=3072)
  wmma_gemm_nt<<<dim3((Tt * DEPTH) / 64, Bsz / 128), 256, 0, stream>>>(
      faugH, selH, Pp, nullptr, Bsz, Tt * DEPTH, FIN);
  // GEMM2: logits = f_aug @ att_W + att_b   (M=4096, N=256, K=3072)
  wmma_gemm_nt<<<dim3(Tt / 64, Bsz / 128), 256, 0, stream>>>(
      faugH, attWtH, Aatt, attb, Bsz, Tt, FIN);

  att_sparsemax_kernel<<<Bsz, 256, 0, stream>>>(Aatt);
  forest_final_kernel<<<Bsz / 4, 256, 0, stream>>>(Pp, Aatt, leaf, out);
}
